// GeneratorBlock_8340826488860
// MI455X (gfx1250) — matrix operand order repaired and checked by reading, not compiled, not run on hardware
//
#include <hip/hip_runtime.h>

// ---------------------------------------------------------------------------
// CDNA5 / gfx1250 implementation of a StyleGAN2 generator block.
// Heavy work (two modulated 3x3 convs, ~232 GFLOP) runs as implicit GEMM on
// v_wmma_f32_16x16x32_bf16 (wave32), with double-buffered LDS staging and
// PT WMMAs issued back-to-back per staged tile. Everything else is fused
// epilogue or tiny elementwise kernels.
// ---------------------------------------------------------------------------

typedef __attribute__((ext_vector_type(16))) __bf16 v16bf;
typedef __attribute__((ext_vector_type(8)))  float  v8f;

#define LIN_SCALE 0.044194173824159216f   /* 1/sqrt(512) */
#define RES_SCALE 0.7071067811865476f     /* sqrt(0.5)   */

__device__ __forceinline__ unsigned short f2bf(float f) {
    unsigned int u = __float_as_uint(f);
    unsigned int lsb = (u >> 16) & 1u;
    u += 0x7fffu + lsb;                   // round to nearest even
    return (unsigned short)(u >> 16);
}
__device__ __forceinline__ float bf2f(unsigned short h) {
    return __uint_as_float(((unsigned int)h) << 16);
}

// ---------------------------------------------------------------------------
// 1) Styles: s = (w * LIN_SCALE) @ A^T + b   for both style blocks.
// ---------------------------------------------------------------------------
__global__ void style_kernel(const float* __restrict__ w,
                             const float* __restrict__ A1w, const float* __restrict__ A1b,
                             const float* __restrict__ A2w, const float* __restrict__ A2b,
                             float* __restrict__ s1, float* __restrict__ s2) {
    int idx = blockIdx.x * blockDim.x + threadIdx.x;       // 0..16383
    int which = idx >> 13;
    int r = idx & 8191;
    int b = r >> 9;
    int i = r & 511;
    const float* Aw = which ? A2w : A1w;
    const float* Ab = which ? A2b : A1b;
    const float* wr = w + b * 512;
    const float* ar = Aw + i * 512;
    float acc = 0.f;
    for (int k = 0; k < 512; ++k) acc += wr[k] * ar[k];
    float res = acc * LIN_SCALE + Ab[i];
    (which ? s2 : s1)[b * 512 + i] = res;
}

// ---------------------------------------------------------------------------
// 2) Demodulation: d[b,o] = rsqrt( sum_i (sum_t w[o,i,t]^2) * s[b,i]^2 + eps )
// ---------------------------------------------------------------------------
__global__ void demod_kernel(const float* __restrict__ s,
                             const float* __restrict__ cw,
                             float* __restrict__ d, int Cout, int Cin) {
    int idx = blockIdx.x * blockDim.x + threadIdx.x;
    if (idx >= 16 * Cout) return;
    int b = idx / Cout;
    int o = idx % Cout;
    float acc = 0.f;
    for (int i = 0; i < Cin; ++i) {
        const float* wp = cw + ((long)o * Cin + i) * 9;
        float ws2 = 0.f;
#pragma unroll
        for (int t = 0; t < 9; ++t) ws2 += wp[t] * wp[t];
        float sv = s[b * Cin + i];
        acc += ws2 * sv * sv;
    }
    d[idx] = rsqrtf(acc + 1e-8f);
}

// ---------------------------------------------------------------------------
// 3) Pack conv weights to bf16 in WMMA A-fragment order so each lane can load
//    its 16x32 bf16 A fragment as two b128 reads.
//    pw index = ((ct*(9*KC) + it)*32 + lane)*16 + j,   it = tap*KC + kc
//    ISA A layout (16-bit, 16x32): M = lane&15,
//                                  K = (j>>3)*16 + (lane>>4)*8 + (j&7)
// ---------------------------------------------------------------------------
__global__ void pack_weights(const float* __restrict__ cw,
                             unsigned short* __restrict__ pw,
                             int Cout, int Cin) {
    long total = (long)Cout * Cin * 9;
    long idx = (long)blockIdx.x * blockDim.x + threadIdx.x;
    if (idx >= total) return;
    int KC = Cin >> 5;
    int j    = (int)(idx & 15);
    int lane = (int)((idx >> 4) & 31);
    long rest = idx >> 9;
    int kc  = (int)(rest % KC); rest /= KC;
    int tap = (int)(rest % 9);
    int ct  = (int)(rest / 9);
    int M = lane & 15;
    int K = ((j >> 3) << 4) + ((lane >> 4) << 3) + (j & 7);
    int cout = ct * 16 + M;
    int cin  = kc * 32 + K;
    float v = cw[((long)cout * Cin + cin) * 9 + tap];
    pw[idx] = f2bf(v);
}

// ---------------------------------------------------------------------------
// 4) Modulate stage-1 input to bf16: xmod1 = x * s1
// ---------------------------------------------------------------------------
__global__ void pack_xmod1(const float* __restrict__ x,
                           const float* __restrict__ s1,
                           unsigned short* __restrict__ xmod) {
    long idx = (long)blockIdx.x * blockDim.x + threadIdx.x;   // 16*512*32*32
    int c = (int)((idx >> 10) & 511);
    int b = (int)(idx >> 19);
    xmod[idx] = f2bf(x[idx] * s1[b * 512 + c]);
}

// ---------------------------------------------------------------------------
// 5) Implicit-GEMM 3x3 conv on WMMA. One block = 256 threads = 8 wave32s.
//    Wave tile: 16 couts x (PT*16) pixels (one full image row).
//    Per (tap, K-chunk of 32 cin): double-buffered LDS stage of a
//    32 x (PT*16) bf16 B tile (one __syncthreads per chunk), one b128-pair
//    A-fragment load reused by PT back-to-back v_wmma_f32_16x16x32_bf16.
//    Epilogue fuses demod, bias, coef*noise, LeakyReLU(0.1).
// ---------------------------------------------------------------------------
template <bool OUT_BF16, int PT>
__global__ void conv3x3_wmma(const unsigned short* __restrict__ xmod,
                             const unsigned short* __restrict__ pw,
                             const float* __restrict__ demod,
                             const float* __restrict__ bias,
                             const float* __restrict__ noise,
                             const float* __restrict__ coef,
                             void* __restrict__ out,
                             int Cin, int Cout, int H, int W) {
    constexpr int COLS = PT * 16;
    __shared__ __align__(16) unsigned short Bs[2][COLS][40];  // [buf][col][K] + pad

    const int tid  = threadIdx.x;
    const int lane = tid & 31;
    const int wid  = tid >> 5;
    const int KC   = Cin >> 5;
    const int NIT  = 9 * KC;
    const int wtiles = W / COLS;

    // pixel tile -> (b, h, w0)
    int pt   = blockIdx.x;
    int wblk = pt % wtiles;
    int tmp  = pt / wtiles;
    int h    = tmp % H;
    int b    = tmp / H;
    int w0   = wblk * COLS;

    int ct = blockIdx.y * 8 + wid;                        // cout tile for this wave

    const v8f vzero = {0.f, 0.f, 0.f, 0.f, 0.f, 0.f, 0.f, 0.f};
    v8f acc[PT];
#pragma unroll
    for (int t = 0; t < PT; ++t) acc[t] = vzero;

    const long planeHW = (long)H * W;
    const long bbase   = (long)b * Cin * planeHW;

    auto stage = [&](int it, int buf) {
        int tap = it / KC;
        int kc  = it - tap * KC;
        int dy  = tap / 3 - 1;
        int dx  = tap % 3 - 1;
        int gy  = h + dy;
        bool rowok   = (gy >= 0) && (gy < H);
        long rowbase = bbase + (long)gy * W;
#pragma unroll
        for (int e = tid; e < COLS * 32; e += 256) {
            int col = e % COLS;                           // COLS is power of 2
            int k   = e / COLS;
            int gx  = w0 + col + dx;
            unsigned short v = 0;
            if (rowok && gx >= 0 && gx < W)
                v = xmod[rowbase + (long)(kc * 32 + k) * planeHW + gx];
            Bs[buf][col][k] = v;
        }
    };

    const uint4* pAct =
        reinterpret_cast<const uint4*>(pw) + (((long)ct * NIT) * 32 + lane) * 2;

    stage(0, 0);
    __syncthreads();

    for (int it = 0; it < NIT; ++it) {
        const int buf = it & 1;
        if (it + 1 < NIT) stage(it + 1, buf ^ 1);         // overlap with WMMAs below

        // ---- A fragment: two b128 loads from pre-swizzled weights
        const uint4* pA = pAct + (long)it * 64;           // 32 lanes * 2 uint4 per it
        union { v16bf v; uint4 q[2]; } ua;
        ua.q[0] = pA[0];
        ua.q[1] = pA[1];
        __builtin_prefetch(pA + 64, 0, 1);                // next fragment -> global_prefetch_b8

        // ---- PT back-to-back WMMAs sharing the A fragment
#pragma unroll
        for (int t = 0; t < PT; ++t) {
            union { v16bf v; uint4 q[2]; } ub;
            const unsigned short* pB = &Bs[buf][t * 16 + (lane & 15)][(lane >> 4) << 3];
            ub.q[0] = *reinterpret_cast<const uint4*>(pB);
            ub.q[1] = *reinterpret_cast<const uint4*>(pB + 16);
            acc[t] = __builtin_amdgcn_wmma_f32_16x16x32_bf16(
                false, ua.v, false, ub.v, (short)0, acc[t], false, false);
        }
        __syncthreads();                                  // one barrier per chunk
    }

    // ---- epilogue: demod + bias + coef*noise + LeakyReLU(0.1)
#pragma unroll
    for (int t = 0; t < PT; ++t) {
        const int w  = w0 + t * 16 + (lane & 15);
        const float ns = coef[0] * noise[((long)b * H + h) * W + w];
#pragma unroll
        for (int r = 0; r < 8; ++r) {
            int cout = ct * 16 + ((lane >> 4) << 3) + r;
            float y = acc[t][r] * demod[b * Cout + cout] + bias[cout] + ns;
            y = (y >= 0.f) ? y : 0.1f * y;
            long o = (((long)b * Cout + cout) * H + h) * W + w;
            if (OUT_BF16) ((unsigned short*)out)[o] = f2bf(y);
            else          ((float*)out)[o] = y;
        }
    }
}

// ---------------------------------------------------------------------------
// 6) Bilinear 2x upsample (half-pixel, like jax.image.resize) of stage-1
//    activation + modulate by s2 -> bf16 xmod2 [16,512,64,64]
// ---------------------------------------------------------------------------
__global__ void upmod_kernel(const unsigned short* __restrict__ act1,
                             const float* __restrict__ s2,
                             unsigned short* __restrict__ xmod2) {
    long idx = (long)blockIdx.x * blockDim.x + threadIdx.x;   // 16*512*64*64
    int X = (int)(idx & 63);
    int Y = (int)((idx >> 6) & 63);
    int c = (int)((idx >> 12) & 511);
    int b = (int)(idx >> 21);
    float fy = Y * 0.5f - 0.25f, fx = X * 0.5f - 0.25f;
    int y0 = (int)floorf(fy), x0 = (int)floorf(fx);
    float wy = fy - y0, wx = fx - x0;
    int y1 = min(max(y0 + 1, 0), 31), x1 = min(max(x0 + 1, 0), 31);
    y0 = min(max(y0, 0), 31); x0 = min(max(x0, 0), 31);
    const unsigned short* p = act1 + (((long)b * 512 + c) << 10);
    float a00 = bf2f(p[(y0 << 5) + x0]);
    float a01 = bf2f(p[(y0 << 5) + x1]);
    float a10 = bf2f(p[(y1 << 5) + x0]);
    float a11 = bf2f(p[(y1 << 5) + x1]);
    float v = (1.f - wy) * ((1.f - wx) * a00 + wx * a01)
            +        wy  * ((1.f - wx) * a10 + wx * a11);
    xmod2[idx] = f2bf(v * s2[b * 512 + c]);
}

// ---------------------------------------------------------------------------
// 7) to_rgb (1x1 conv 256->3) + bilinear-upsampled rgb residual, * sqrt(0.5)
// ---------------------------------------------------------------------------
__global__ void rgb_kernel(const float* __restrict__ xout,
                           const float* __restrict__ rgbw,
                           const float* __restrict__ rgbb,
                           const float* __restrict__ rgb_in,
                           float* __restrict__ out) {
    int idx = blockIdx.x * blockDim.x + threadIdx.x;          // 16*64*64
    int X = idx & 63;
    int Y = (idx >> 6) & 63;
    int b = idx >> 12;

    float acc0 = rgbb[0], acc1 = rgbb[1], acc2 = rgbb[2];
    const float* xp = xout + ((long)b * 256) * 4096 + (Y << 6) + X;
    for (int c = 0; c < 256; ++c) {
        float xv = xp[(long)c * 4096];
        acc0 += rgbw[c] * xv;
        acc1 += rgbw[256 + c] * xv;
        acc2 += rgbw[512 + c] * xv;
    }

    float fy = Y * 0.5f - 0.25f, fx = X * 0.5f - 0.25f;
    int y0 = (int)floorf(fy), x0 = (int)floorf(fx);
    float wy = fy - y0, wx = fx - x0;
    int y1 = min(max(y0 + 1, 0), 31), x1 = min(max(x0 + 1, 0), 31);
    y0 = min(max(y0, 0), 31); x0 = min(max(x0, 0), 31);

    float acc[3] = {acc0, acc1, acc2};
#pragma unroll
    for (int o = 0; o < 3; ++o) {
        const float* rp = rgb_in + (((long)b * 3 + o) << 10);
        float a00 = rp[(y0 << 5) + x0], a01 = rp[(y0 << 5) + x1];
        float a10 = rp[(y1 << 5) + x0], a11 = rp[(y1 << 5) + x1];
        float up = (1.f - wy) * ((1.f - wx) * a00 + wx * a01)
                 +        wy  * ((1.f - wx) * a10 + wx * a11);
        out[(((long)b * 3 + o) << 12) + (Y << 6) + X] = (acc[o] + up) * RES_SCALE;
    }
}

// ---------------------------------------------------------------------------
extern "C" void kernel_launch(void* const* d_in, const int* in_sizes, int n_in,
                              void* d_out, int out_size, void* d_ws, size_t ws_size,
                              hipStream_t stream) {
    const float* x       = (const float*)d_in[0];
    const float* w       = (const float*)d_in[1];
    const float* rgb     = (const float*)d_in[2];
    const float* noise1  = (const float*)d_in[3];
    const float* noise2  = (const float*)d_in[4];
    const float* A1w     = (const float*)d_in[5];
    const float* A1b     = (const float*)d_in[6];
    const float* conv1w  = (const float*)d_in[7];
    const float* conv1b  = (const float*)d_in[8];
    const float* coef1   = (const float*)d_in[9];
    const float* A2w     = (const float*)d_in[10];
    const float* A2b     = (const float*)d_in[11];
    const float* conv2w  = (const float*)d_in[12];
    const float* conv2b  = (const float*)d_in[13];
    const float* coef2   = (const float*)d_in[14];
    const float* rgbw    = (const float*)d_in[15];
    const float* rgbb    = (const float*)d_in[16];

    float* xout_part = (float*)d_out;                      // [16,256,64,64]
    float* rgb_part  = xout_part + (long)16 * 256 * 64 * 64;

    // ---- carve workspace (256B aligned slabs)
    char* p = (char*)d_ws;
    auto carve = [&](size_t bytes) -> void* {
        void* r = (void*)p;
        p += (bytes + 255) & ~(size_t)255;
        return r;
    };
    float*          s1    = (float*)carve(16 * 512 * 4);
    float*          s2    = (float*)carve(16 * 512 * 4);
    float*          dd1   = (float*)carve(16 * 512 * 4);
    float*          dd2   = (float*)carve(16 * 256 * 4);
    unsigned short* pw1   = (unsigned short*)carve((size_t)512 * 512 * 9 * 2);
    unsigned short* pw2   = (unsigned short*)carve((size_t)256 * 512 * 9 * 2);
    unsigned short* xmod1 = (unsigned short*)carve((size_t)16 * 512 * 32 * 32 * 2);
    unsigned short* act1  = (unsigned short*)carve((size_t)16 * 512 * 32 * 32 * 2);
    unsigned short* xmod2 = (unsigned short*)carve((size_t)16 * 512 * 64 * 64 * 2);

    // 1) styles
    style_kernel<<<64, 256, 0, stream>>>(w, A1w, A1b, A2w, A2b, s1, s2);
    // 2) demod factors
    demod_kernel<<<(16 * 512 + 255) / 256, 256, 0, stream>>>(s1, conv1w, dd1, 512, 512);
    demod_kernel<<<(16 * 256 + 255) / 256, 256, 0, stream>>>(s2, conv2w, dd2, 256, 512);
    // 3) weight packing (A-fragment swizzle)
    pack_weights<<<(int)(((long)512 * 512 * 9 + 255) / 256), 256, 0, stream>>>(conv1w, pw1, 512, 512);
    pack_weights<<<(int)(((long)256 * 512 * 9 + 255) / 256), 256, 0, stream>>>(conv2w, pw2, 256, 512);
    // 4) modulate input
    pack_xmod1<<<(int)((long)16 * 512 * 32 * 32 / 256), 256, 0, stream>>>(x, s1, xmod1);
    // 5) conv1: 512->512 @ 32x32 (row tile = 32 px, PT=2), bf16 out
    {
        dim3 grid(16 * 32, 512 / 128);
        conv3x3_wmma<true, 2><<<grid, 256, 0, stream>>>(xmod1, pw1, dd1, conv1b, noise1,
                                                        coef1, (void*)act1, 512, 512, 32, 32);
    }
    // 6) upsample + modulate for stage 2
    upmod_kernel<<<(int)((long)16 * 512 * 64 * 64 / 256), 256, 0, stream>>>(act1, s2, xmod2);
    // 7) conv2: 512->256 @ 64x64 (row tile = 64 px, PT=4), f32 out -> d_out
    {
        dim3 grid(16 * 64, 256 / 128);
        conv3x3_wmma<false, 4><<<grid, 256, 0, stream>>>(xmod2, pw2, dd2, conv2b, noise2,
                                                         coef2, (void*)xout_part, 512, 256, 64, 64);
    }
    // 8) to_rgb + residual
    rgb_kernel<<<16 * 64 * 64 / 256, 256, 0, stream>>>(xout_part, rgbw, rgbb, rgb, rgb_part);
}
